// MoERouter_28406913695888
// MI455X (gfx1250) — matrix operand order repaired and checked by reading, not compiled, not run on hardware
//
#include <hip/hip_runtime.h>

// ---------------------------------------------------------------------------
// MoE router for MI455X (gfx1250, wave32, WMMA).
// logits = x @ W^T  (N=16384 tokens, D=2048, E=64), top-8 + softmax + aux loss.
// GEMM via v_wmma_f32_16x16x32_bf16. Each wave owns 32 tokens x 64 experts
// (2 A fragments x 4 B tiles = 8 WMMAs per K-step) so each B fragment fetch
// is amortized over twice the tokens (x HBM traffic is the roofline floor:
// 134 MB @ 23.3 TB/s ~= 5.8 us).
// ---------------------------------------------------------------------------

typedef __attribute__((ext_vector_type(16))) __bf16 v16bf;
typedef __attribute__((ext_vector_type(8)))  float  v8f;

#define D_DIM            2048
#define E_DIM            64
#define N_TOK            16384      // B*S = 4*4096
#define TOPK             8
#define KB_COUNT         64         // D / 32
#define WAVES_PER_BLOCK  4
#define TOKENS_PER_WAVE  32         // two 16-row M tiles
#define TOKENS_PER_BLOCK (WAVES_PER_BLOCK * TOKENS_PER_WAVE)   // 128
#define BLOCK_THREADS    (WAVES_PER_BLOCK * 32)                // 128
#define LROW             (E_DIM + 1)  // LDS row stride (bank-conflict padding)

// B-fragment records: [tile(4)][kb(64)][lane(32)] x 16 bf16 (32 bytes each).
#define BPACK_RECORDS    (4 * KB_COUNT * 32)
#define BPACK_BYTES      (BPACK_RECORDS * 16 * 2)

// ---------------------------------------------------------------------------
// Kernel 1: pack W_gate (E x D fp32, row major) into bf16 WMMA B-fragments.
// 16x16x32 bf16 WMMA B (32x16) layout per lane:
//   lanes 0-15 : N = lane,    K = {0..7, 16..23}
//   lanes 16-31: N = lane-16, K = {8..15, 24..31}
// B[k][n] = W[n_global][kbase + k]  (logits[m,e] = sum_d x[m,d]*W[e,d]).
// ---------------------------------------------------------------------------
__global__ void pack_w_kernel(const float* __restrict__ W, __bf16* __restrict__ Bp) {
    int id = blockIdx.x * blockDim.x + threadIdx.x;      // 0 .. 8191
    if (id >= BPACK_RECORDS) return;
    int lane = id & 31;
    int kb   = (id >> 5) & (KB_COUNT - 1);
    int t    = id >> 11;                                  // expert tile 0..3
    int e    = t * 16 + (lane & 15);
    int kbase = kb * 32 + ((lane >> 4) * 8);
    const float* src = W + (size_t)e * D_DIM + kbase;
    __bf16* dst = Bp + (size_t)id * 16;
#pragma unroll
    for (int j = 0; j < 16; ++j)
        dst[j] = (__bf16)src[(j & 7) + (j >> 3) * 16];
}

// ---------------------------------------------------------------------------
// Kernel 2: zero the global f/P accumulators (graph-capture safe init).
// ---------------------------------------------------------------------------
__global__ void zero_acc_kernel(float* __restrict__ acc) {
    if (threadIdx.x < 2 * E_DIM) acc[threadIdx.x] = 0.0f;
}

// ---------------------------------------------------------------------------
// A-fragment helper: build 16x32 bf16 A fragment for 16 rows starting at
// x + row*D + half*8 + kb*32.  Per-lane K pattern:
//   lanes 0-15 : K = {0..7, 16..23},  lanes 16-31: K = {8..15, 24..31}
// ---------------------------------------------------------------------------
__device__ __forceinline__ v16bf load_a_frag(const float* __restrict__ xp) {
    float4 f0 = *(const float4*)(xp + 0);
    float4 f1 = *(const float4*)(xp + 4);
    float4 f2 = *(const float4*)(xp + 16);
    float4 f3 = *(const float4*)(xp + 20);
    v16bf a;
    a[0]  = (__bf16)f0.x; a[1]  = (__bf16)f0.y; a[2]  = (__bf16)f0.z; a[3]  = (__bf16)f0.w;
    a[4]  = (__bf16)f1.x; a[5]  = (__bf16)f1.y; a[6]  = (__bf16)f1.z; a[7]  = (__bf16)f1.w;
    a[8]  = (__bf16)f2.x; a[9]  = (__bf16)f2.y; a[10] = (__bf16)f2.z; a[11] = (__bf16)f2.w;
    a[12] = (__bf16)f3.x; a[13] = (__bf16)f3.y; a[14] = (__bf16)f3.z; a[15] = (__bf16)f3.w;
    return a;
}

// ---------------------------------------------------------------------------
// Kernel 3: main router. One wave -> 32 tokens x 64 experts.
// ---------------------------------------------------------------------------
__global__ void __launch_bounds__(BLOCK_THREADS)
router_kernel(const float* __restrict__ x,
              const __bf16* __restrict__ Bpack,
              float* __restrict__ out_w,
              int*   __restrict__ out_idx,
              float* __restrict__ f_sum,
              float* __restrict__ P_sum) {
    __shared__ float lds_logits[TOKENS_PER_BLOCK][LROW];
    __shared__ float f_loc[E_DIM];
    __shared__ float P_loc[E_DIM];

    const int tid  = threadIdx.x;
    const int wave = tid >> 5;
    const int lane = tid & 31;
    const int half = lane >> 4;      // 0: lanes 0-15, 1: lanes 16-31
    const int l15  = lane & 15;

    if (tid < E_DIM) { f_loc[tid] = 0.0f; P_loc[tid] = 0.0f; }
    __syncthreads();

    // ---------------- GEMM phase: logits[32 tokens][64 experts] ------------
    const int token_base = (blockIdx.x * WAVES_PER_BLOCK + wave) * TOKENS_PER_WAVE;

    v8f acc[4][2];
#pragma unroll
    for (int t = 0; t < 4; ++t) { acc[t][0] = (v8f){}; acc[t][1] = (v8f){}; }

    const float* xrow0 = x + (size_t)(token_base + l15)      * D_DIM + half * 8;
    const float* xrow1 = x + (size_t)(token_base + 16 + l15) * D_DIM + half * 8;
    const v16bf* bptr  = (const v16bf*)Bpack;

#pragma unroll 2
    for (int kb = 0; kb < KB_COUNT; ++kb) {
        v16bf a0 = load_a_frag(xrow0 + kb * 32);
        v16bf a1 = load_a_frag(xrow1 + kb * 32);

        v16bf b0 = bptr[(0 * KB_COUNT + kb) * 32 + lane];
        v16bf b1 = bptr[(1 * KB_COUNT + kb) * 32 + lane];
        v16bf b2 = bptr[(2 * KB_COUNT + kb) * 32 + lane];
        v16bf b3 = bptr[(3 * KB_COUNT + kb) * 32 + lane];

        acc[0][0] = __builtin_amdgcn_wmma_f32_16x16x32_bf16(false, a0, false, b0, (short)0, acc[0][0], false, false);
        acc[0][1] = __builtin_amdgcn_wmma_f32_16x16x32_bf16(false, a1, false, b0, (short)0, acc[0][1], false, false);
        acc[1][0] = __builtin_amdgcn_wmma_f32_16x16x32_bf16(false, a0, false, b1, (short)0, acc[1][0], false, false);
        acc[1][1] = __builtin_amdgcn_wmma_f32_16x16x32_bf16(false, a1, false, b1, (short)0, acc[1][1], false, false);
        acc[2][0] = __builtin_amdgcn_wmma_f32_16x16x32_bf16(false, a0, false, b2, (short)0, acc[2][0], false, false);
        acc[2][1] = __builtin_amdgcn_wmma_f32_16x16x32_bf16(false, a1, false, b2, (short)0, acc[2][1], false, false);
        acc[3][0] = __builtin_amdgcn_wmma_f32_16x16x32_bf16(false, a0, false, b3, (short)0, acc[3][0], false, false);
        acc[3][1] = __builtin_amdgcn_wmma_f32_16x16x32_bf16(false, a1, false, b3, (short)0, acc[3][1], false, false);
    }

    // Spill logits to LDS. C/D f32 layout per tile: VGPR v: lanes 0-15 ->
    // (M=v, N=lane), lanes 16-31 -> (M=v+8, N=lane-16).
    {
        const int rbase = wave * TOKENS_PER_WAVE;
#pragma unroll
        for (int mt = 0; mt < 2; ++mt) {
#pragma unroll
            for (int v = 0; v < 8; ++v) {
                int r = rbase + mt * 16 + v + 8 * half;
                lds_logits[r][ 0 + l15] = acc[0][mt][v];
                lds_logits[r][16 + l15] = acc[1][mt][v];
                lds_logits[r][32 + l15] = acc[2][mt][v];
                lds_logits[r][48 + l15] = acc[3][mt][v];
            }
        }
    }
    __syncthreads();

    // ---------------- top-k + softmax + aux-loss stats ---------------------
    // All 128 threads: one token each.
    {
        const float* L = &lds_logits[tid][0];
        const int token = blockIdx.x * TOKENS_PER_BLOCK + tid;

        // full-softmax stats over all 64 experts (for P)
        float mx = L[0];
#pragma unroll
        for (int e = 1; e < E_DIM; ++e) mx = fmaxf(mx, L[e]);
        float denom = 0.0f;
#pragma unroll
        for (int e = 0; e < E_DIM; ++e) denom += __expf(L[e] - mx);
        float inv_denom = 1.0f / denom;
#pragma unroll
        for (int e = 0; e < E_DIM; ++e)
            atomicAdd(&P_loc[e], __expf(L[e] - mx) * inv_denom);

        // top-8 by repeated masked max
        unsigned long long chosen = 0ull;
        float tv[TOPK];
        int   tix[TOPK];
#pragma unroll
        for (int j = 0; j < TOPK; ++j) {
            float best = -3.4e38f;
            int   bi   = 0;
            for (int e = 0; e < E_DIM; ++e) {
                float v = L[e];
                bool used = (chosen >> e) & 1ull;
                if (!used && v > best) { best = v; bi = e; }
            }
            chosen |= (1ull << bi);
            tv[j] = best;
            tix[j] = bi;
        }

        // softmax over the selected logits (tv[0] is the max)
        float w8[TOPK];
        float s = 0.0f;
#pragma unroll
        for (int j = 0; j < TOPK; ++j) { w8[j] = __expf(tv[j] - tv[0]); s += w8[j]; }
        float invs = 1.0f / s;
#pragma unroll
        for (int j = 0; j < TOPK; ++j) {
            out_w[(size_t)token * TOPK + j]   = w8[j] * invs;
            out_idx[(size_t)token * TOPK + j] = tix[j];
            atomicAdd(&f_loc[tix[j]], 1.0f);
        }
    }
    __syncthreads();

    if (tid < E_DIM) {
        atomicAdd(&f_sum[tid], f_loc[tid]);
        atomicAdd(&P_sum[tid], P_loc[tid]);
    }
}

// ---------------------------------------------------------------------------
// Kernel 4: aux = E * sum_e (f_e/N) * (P_e/N) * 0.01
// ---------------------------------------------------------------------------
__global__ void finalize_kernel(const float* __restrict__ f_sum,
                                const float* __restrict__ P_sum,
                                float* __restrict__ aux_out) {
    __shared__ float red[E_DIM];
    int e = threadIdx.x;
    float fv = f_sum[e] * (1.0f / (float)N_TOK);
    float pv = P_sum[e] * (1.0f / (float)N_TOK);
    red[e] = fv * pv;
    __syncthreads();
    if (e == 0) {
        float s = 0.0f;
#pragma unroll
        for (int i = 0; i < E_DIM; ++i) s += red[i];
        *aux_out = (float)E_DIM * s * 0.01f;
    }
}

// ---------------------------------------------------------------------------
// Launch
// ---------------------------------------------------------------------------
extern "C" void kernel_launch(void* const* d_in, const int* in_sizes, int n_in,
                              void* d_out, int out_size, void* d_ws, size_t ws_size,
                              hipStream_t stream) {
    (void)in_sizes; (void)n_in; (void)out_size; (void)ws_size;

    const float* x = (const float*)d_in[0];   // (4, 4096, 2048) fp32
    const float* W = (const float*)d_in[1];   // (64, 2048) fp32

    float* out   = (float*)d_out;
    float* out_w = out;                                   // (N, 8) weights
    int*   out_i = (int*)(out + (size_t)N_TOK * TOPK);    // (N, 8) indices
    float* aux   = out + 2 * (size_t)N_TOK * TOPK;        // scalar

    __bf16* Bpack = (__bf16*)d_ws;
    float*  f_sum = (float*)((char*)d_ws + BPACK_BYTES);
    float*  P_sum = f_sum + E_DIM;

    pack_w_kernel<<<BPACK_RECORDS / 256, 256, 0, stream>>>(W, Bpack);
    zero_acc_kernel<<<1, 2 * E_DIM, 0, stream>>>(f_sum);
    router_kernel<<<N_TOK / TOKENS_PER_BLOCK, BLOCK_THREADS, 0, stream>>>(
        x, Bpack, out_w, out_i, f_sum, P_sum);
    finalize_kernel<<<1, E_DIM, 0, stream>>>(f_sum, P_sum, aux);
}
